// CrossAttention_41695542509808
// MI455X (gfx1250) — matrix operand order repaired and checked by reading, not compiled, not run on hardware
//
#include <hip/hip_runtime.h>

typedef _Float16 f16;
typedef __attribute__((ext_vector_type(16))) _Float16 v16h;
typedef __attribute__((ext_vector_type(8)))  _Float16 v8h;
typedef __attribute__((ext_vector_type(8)))  float    v8f;

#define NB 16
#define QL 4096
#define CL 77
#define DC 768
#define DM 512
#define NH 8
#define SH 64

// workspace layout (in halfs)
#define WQB_OFF 0
#define WQB_SZ  (512 * 512)            // B[512][512] frags (Wq, n = h*64+s)
#define WOB_OFF (WQB_OFF + WQB_SZ)
#define WOB_SZ  (512 * 512)            // B[512][512] frags (Wo)
#define KTB_OFF (WOB_OFF + WOB_SZ)
#define KTB_SZ  (NB * NH * 10 * 512)   // per (n,h): B[64][80] -> 10 frags
#define VB_OFF  (KTB_OFF + KTB_SZ)
#define VB_SZ   (NB * NH * 12 * 512)   // per (n,h): B[96][64] -> 12 frags

// dynamic LDS layout (bytes)
#define QBUF_B  0                      // [64][512] f16 q/attn-out  (65536)
#define BSTG_B  65536                  // 2 x 32KB B-chunk buffers  (65536)
#define QCHK_B  (BSTG_B + 65536)       // 2 x [64][32] f16          (8192)
#define WSC_B   (QCHK_B + 8192)        // per-wave 2x[16][96] f16   (49152)
#define LDS_TOTAL (WSC_B + 49152)      // 188416

static __device__ inline v8f wmma16(v16h a, v16h b, v8f c) {
  return __builtin_amdgcn_wmma_f32_16x16x32_f16(false, a, false, b, (short)0, c,
                                                false, false);
}
static __device__ inline v16h cat8(v8h lo, v8h hi) {
  return __builtin_shufflevector(lo, hi, 0, 1, 2, 3, 4, 5, 6, 7, 8, 9, 10, 11,
                                 12, 13, 14, 15);
}

// ---- CDNA5 async global->LDS copy (ASYNCcnt-tracked, no data VGPRs) ----
static __device__ inline void async_b128(unsigned ldsoff, const void* gaddr) {
  asm volatile("global_load_async_to_lds_b128 %0, %1, off" ::"v"(ldsoff),
               "v"((unsigned long long)(size_t)gaddr)
               : "memory");
}
static __device__ inline void wait_async0() {
  asm volatile("s_wait_asynccnt 0" ::: "memory");
}
// each thread copies 128 contiguous bytes (8 x b128) of a 32KB chunk
static __device__ inline void stage_async(const f16* src, f16* dstLds, int t) {
  unsigned base = (unsigned)(size_t)(dstLds + t * 64);  // LDS offset = addr[31:0]
  const f16* s = src + t * 64;
#pragma unroll
  for (int j = 0; j < 8; ++j) async_b128(base + j * 16, s + j * 8);
}

// ---------------------------------------------------------------------------
// Kernel 1: repack Wq / Wo (f32 -> f16) into per-fragment B layout.
// Both are flat [512][512] row-major (Wq's [D,H,S] flattens to [d][h*64+s]).
// B frag mapping: f = kc*32 + nt, lane = n + 16*(kl>=16), element e = kl&15.
// ---------------------------------------------------------------------------
__global__ __launch_bounds__(256) void prep_weights(const float* __restrict__ Wq,
                                                    const float* __restrict__ Wo,
                                                    f16* __restrict__ ws) {
  int tid = blockIdx.x * 256 + threadIdx.x;   // 0 .. 524287
  int idx = tid & (WQB_SZ - 1);
  int e = idx & 15;
  int lane = (idx >> 4) & 31;
  int kl = e + (lane & 16);
  int nl = lane & 15;
  int f = idx >> 9;                           // 0..511
  int k = (f >> 5) * 32 + kl;
  int nn = (f & 31) * 16 + nl;
  const float* W = (tid < WQB_SZ) ? Wq : Wo;
  ws[tid] = (f16)W[k * 512 + nn];
}

// ---------------------------------------------------------------------------
// Kernel 2: K/V projections with WMMA; emit kT / v in B-fragment layout.
// One block per (n,h); 10 waves = 5 m-tiles x {K-matrix, V-matrix}.
// ---------------------------------------------------------------------------
__global__ __launch_bounds__(320) void prep_kv(const float* __restrict__ ctx,
                                               const float* __restrict__ Wk,
                                               const float* __restrict__ Wv,
                                               f16* __restrict__ ws) {
  __shared__ f16 bstage[2][4][512];
  int nh = blockIdx.x;                 // n*8 + h
  int n = nh >> 3, h = nh & 7;
  int t = threadIdx.x;
  int wave = t >> 5, lane = t & 31;
  int lo = lane & 15, hi = lane >> 4;

  f16* kTB = ws + KTB_OFF + nh * (10 * 512);
  f16* vB  = ws + VB_OFF  + nh * (12 * 512);

  // zero-fill v pad rows c=80..95 (frags 8..11, lanes 16..31, all e) — disjoint
  // from the scatter region below, so no intra-block ordering needed.
  if (t < 64) {
    int f = 8 + (t >> 4);
    int lz = 16 + (t & 15);
    v8h z = {};
    *(v8h*)(vB + (f * 32 + lz) * 16)     = z;
    *(v8h*)(vB + (f * 32 + lz) * 16 + 8) = z;
  }

  int mat = (wave >= 5) ? 1 : 0;
  int mt  = mat ? (wave - 5) : wave;

  v8f acc[4] = {};

  for (int kc = 0; kc < 24; ++kc) {
    __syncthreads();
    if (t < 256) {  // stage Wk/Wv chunk [32][64] into fragment-ordered LDS
      int ms = t >> 7, fs = (t >> 5) & 3, ls = t & 31;
      const float* Ws = ms ? Wv : Wk;
      int d0 = kc * 32 + (ls & 16);
      int s  = fs * 16 + (ls & 15);
      f16* dst = &bstage[ms][fs][ls * 16];
#pragma unroll
      for (int e2 = 0; e2 < 16; ++e2)
        dst[e2] = (f16)Ws[((d0 + e2) * NH + h) * SH + s];
    }
    __syncthreads();
    // A fragment: context rows (m = c), zero-padded past C=77
    int c = mt * 16 + lo;
    v16h af;
    if (c < CL) {
      const float4* p =
          (const float4*)(ctx + (n * CL + c) * DC + kc * 32 + hi * 16);
      float4 x0 = p[0], x1 = p[1], y0 = p[2], y1 = p[3];
      af[0] = (f16)x0.x; af[1] = (f16)x0.y; af[2] = (f16)x0.z; af[3] = (f16)x0.w;
      af[4] = (f16)x1.x; af[5] = (f16)x1.y; af[6] = (f16)x1.z; af[7] = (f16)x1.w;
      af[8] = (f16)y0.x; af[9] = (f16)y0.y; af[10] = (f16)y0.z; af[11] = (f16)y0.w;
      af[12] = (f16)y1.x; af[13] = (f16)y1.y; af[14] = (f16)y1.z; af[15] = (f16)y1.w;
    } else {
#pragma unroll
      for (int e2 = 0; e2 < 16; ++e2) af[e2] = (f16)0.f;
    }
#pragma unroll
    for (int nt = 0; nt < 4; ++nt)
      acc[nt] = wmma16(af, *(const v16h*)&bstage[mat][nt][lane * 16], acc[nt]);
  }

  // scatter D tiles (m=c, n=s) into B-fragment layout for the attention kernel
#pragma unroll
  for (int nt = 0; nt < 4; ++nt) {
#pragma unroll
    for (int r = 0; r < 8; ++r) {
      int c = mt * 16 + r + hi * 8;
      int s = nt * 16 + lo;
      f16 val = (f16)acc[nt][r];
      if (mat == 0) {
        // kT: B[k=s][n=c], 5 n-tiles
        int ks = s >> 5, kl2 = s & 31;
        int f = ks * 5 + (c >> 4);
        int ln = (c & 15) + (kl2 & 16);
        kTB[(f * 32 + ln) * 16 + (kl2 & 15)] = val;
      } else {
        // v: B[k=c][n=s], 4 n-tiles
        int ks = c >> 5, kl2 = c & 31;
        int f = ks * 4 + (s >> 4);
        int ln = (s & 15) + (kl2 & 16);
        vB[(f * 32 + ln) * 16 + (kl2 & 15)] = val;
      }
    }
  }
}

// ---------------------------------------------------------------------------
// Kernel 3: fused attention per (batch, 64-row query tile).
// ---------------------------------------------------------------------------
__global__ __launch_bounds__(256) void attn_main(const float* __restrict__ query,
                                                 const float* __restrict__ bo,
                                                 const f16* __restrict__ ws,
                                                 float* __restrict__ out) {
  extern __shared__ char smem[];
  f16* qbuf  = (f16*)(smem + QBUF_B);            // [64][512]: q, then attn-out
  f16* bstg0 = (f16*)(smem + BSTG_B);            // B-chunk buffer 0
  f16* bstg1 = (f16*)(smem + BSTG_B + 32768);    // B-chunk buffer 1
  f16* qchk0 = (f16*)(smem + QCHK_B);            // query chunk buffer 0
  f16* qchk1 = (f16*)(smem + QCHK_B + 4096);     // query chunk buffer 1
  int t = threadIdx.x, wave = t >> 5, lane = t & 31;
  int lo = lane & 15, hi = lane >> 4;
  f16* wscw = (f16*)(smem + WSC_B) + wave * 3072;  // 2 x [16][96]

  int n = blockIdx.y;
  int q0 = blockIdx.x * 64;
  int h = wave;
  int mtw = wave & 3;        // stage-1/3 m-tile
  int nhf = wave >> 2;       // stage-1/3 n-tile half (16 tiles each)

  const f16* wqB = ws + WQB_OFF;
  const f16* woB = ws + WOB_OFF;
  const f16* kTB = ws + KTB_OFF + (n * NH + h) * (10 * 512);
  const f16* vB  = ws + VB_OFF  + (n * NH + h) * (12 * 512);

  int qrow = t >> 2, qc8 = (t & 3) * 8;  // query-chunk staging assignment
  const float* qsrc = query + (size_t)(n * QL + q0 + qrow) * DM + qc8;

  // -------- Stage 1: q[64,512] = query_tile[64,512] x Wq[512,512] --------
  {
    v8f acc[16] = {};
    stage_async(wqB, bstg0, t);               // async B chunk for kc=0
    float4 x = ((const float4*)qsrc)[0];      // query chunk kc=0
    float4 y = ((const float4*)qsrc)[1];
    for (int kc = 0; kc < 16; ++kc) {
      f16* bs = (kc & 1) ? bstg1 : bstg0;
      f16* qc = (kc & 1) ? qchk1 : qchk0;
      {  // convert + store this kc's query chunk (loaded last iteration)
        v8h qv;
        qv[0] = (f16)x.x; qv[1] = (f16)x.y; qv[2] = (f16)x.z; qv[3] = (f16)x.w;
        qv[4] = (f16)y.x; qv[5] = (f16)y.y; qv[6] = (f16)y.z; qv[7] = (f16)y.w;
        *(v8h*)(qc + qrow * 32 + qc8) = qv;
      }
      wait_async0();
      __syncthreads();
      if (kc < 15) {  // overlap next B chunk copy + next query loads w/ compute
        stage_async(wqB + (kc + 1) * 16384, (kc & 1) ? bstg0 : bstg1, t);
        const float4* p = (const float4*)(qsrc + (kc + 1) * 32);
        x = p[0];
        y = p[1];
      }
      const f16* ap = qc + (mtw * 16 + lo) * 32 + hi * 16;
      v16h a = cat8(*(const v8h*)ap, *(const v8h*)(ap + 8));
      const f16* bbase = bs + (nhf * 16 * 32 + lane) * 16;
      v16h bf = *(const v16h*)bbase;
#pragma unroll
      for (int nt = 0; nt < 15; ++nt) {  // keep one B frag in flight
        v16h bn = *(const v16h*)(bbase + (nt + 1) * 512);
        acc[nt] = wmma16(a, bf, acc[nt]);
        bf = bn;
      }
      acc[15] = wmma16(a, bf, acc[15]);
    }
#pragma unroll
    for (int nt = 0; nt < 16; ++nt)
#pragma unroll
      for (int r = 0; r < 8; ++r)
        qbuf[(mtw * 16 + r + hi * 8) * 512 + (nhf * 16 + nt) * 16 + lo] =
            (f16)acc[nt][r];
  }
  __syncthreads();

  // -------- Stage 2: per-head attention, two m-tiles per pass --------
  for (int mtp = 0; mtp < 2; ++mtp) {
    v16h qa[2][2];
#pragma unroll
    for (int m2 = 0; m2 < 2; ++m2) {
      const f16* p =
          qbuf + ((mtp * 2 + m2) * 16 + lo) * 512 + h * 64 + hi * 16;
#pragma unroll
      for (int kc = 0; kc < 2; ++kc)
        qa[m2][kc] = cat8(*(const v8h*)(p + kc * 32),
                          *(const v8h*)(p + kc * 32 + 8));
    }
    v8f sc[2][5];
#pragma unroll
    for (int nt = 0; nt < 5; ++nt) {
      v8f s0 = {}, s1 = {};
#pragma unroll
      for (int kc = 0; kc < 2; ++kc) {
        v16h bf = *(const v16h*)(kTB + ((kc * 5 + nt) * 32 + lane) * 16);
        s0 = wmma16(qa[0][kc], bf, s0);
        s1 = wmma16(qa[1][kc], bf, s1);
      }
      sc[0][nt] = s0;
      sc[1][nt] = s1;
    }
    // softmax over c; each row lives in 16 lanes (one half-wave) x 5 tiles
    bool tailok = lo < 13;  // c = 64 + lo valid iff < 77
#pragma unroll
    for (int m2 = 0; m2 < 2; ++m2) {
      float mx[8], sum[8];
#pragma unroll
      for (int r = 0; r < 8; ++r) {
#pragma unroll
        for (int nt = 0; nt < 5; ++nt) sc[m2][nt][r] *= 0.125f;  // 1/sqrt(64)
        float m = fmaxf(fmaxf(sc[m2][0][r], sc[m2][1][r]),
                        fmaxf(sc[m2][2][r], sc[m2][3][r]));
        mx[r] = fmaxf(m, tailok ? sc[m2][4][r] : -1e30f);
      }
#pragma unroll
      for (int msk = 1; msk <= 8; msk <<= 1)
#pragma unroll
        for (int r = 0; r < 8; ++r)
          mx[r] = fmaxf(mx[r], __shfl_xor(mx[r], msk, 32));
#pragma unroll
      for (int r = 0; r < 8; ++r) sum[r] = 0.f;
#pragma unroll
      for (int nt = 0; nt < 5; ++nt)
#pragma unroll
        for (int r = 0; r < 8; ++r) {
          float ev = __expf(sc[m2][nt][r] - mx[r]);
          if (nt == 4 && !tailok) ev = 0.f;
          sc[m2][nt][r] = ev;
          sum[r] += ev;
        }
#pragma unroll
      for (int msk = 1; msk <= 8; msk <<= 1)
#pragma unroll
        for (int r = 0; r < 8; ++r) sum[r] += __shfl_xor(sum[r], msk, 32);
#pragma unroll
      for (int r = 0; r < 8; ++r) sum[r] = 1.f / sum[r];
      // weights -> per-wave LDS scratch [16][96] (cols >= 77 zero)
      f16* wsc = wscw + m2 * 1536;
#pragma unroll
      for (int nt = 0; nt < 5; ++nt)
#pragma unroll
        for (int r = 0; r < 8; ++r)
          wsc[(r + hi * 8) * 96 + nt * 16 + lo] = (f16)(sc[m2][nt][r] * sum[r]);
      v8h z = {};
      *(v8h*)(wsc + lo * 96 + 80 + hi * 8) = z;  // pad cols 80..95
    }
    __builtin_amdgcn_wave_barrier();
    v16h wa[2][3];
#pragma unroll
    for (int m2 = 0; m2 < 2; ++m2) {
      const f16* p = wscw + m2 * 1536 + lo * 96 + hi * 16;
#pragma unroll
      for (int kc = 0; kc < 3; ++kc)
        wa[m2][kc] = cat8(*(const v8h*)(p + kc * 32),
                          *(const v8h*)(p + kc * 32 + 8));
    }
    v8f oa[2][4] = {};
#pragma unroll
    for (int nt = 0; nt < 4; ++nt)
#pragma unroll
      for (int kc = 0; kc < 3; ++kc) {
        v16h bf = *(const v16h*)(vB + ((kc * 4 + nt) * 32 + lane) * 16);
        oa[0][nt] = wmma16(wa[0][kc], bf, oa[0][nt]);
        oa[1][nt] = wmma16(wa[1][kc], bf, oa[1][nt]);
      }
#pragma unroll
    for (int m2 = 0; m2 < 2; ++m2)
#pragma unroll
      for (int nt = 0; nt < 4; ++nt)
#pragma unroll
        for (int r = 0; r < 8; ++r)
          qbuf[((mtp * 2 + m2) * 16 + r + hi * 8) * 512 + h * 64 + nt * 16 +
               lo] = (f16)oa[m2][nt][r];
  }

  // -------- Stage 3: out[64,512] = attn[64,512] x Wo[512,512] + bo --------
  {
    v8f acc[16] = {};
    stage_async(woB, bstg0, t);   // bstg free since stage 1; qbuf races guarded
    for (int kc = 0; kc < 16; ++kc) {
      f16* bs = (kc & 1) ? bstg1 : bstg0;
      wait_async0();
      __syncthreads();
      if (kc < 15)
        stage_async(woB + (kc + 1) * 16384, (kc & 1) ? bstg0 : bstg1, t);
      const f16* ap = qbuf + (mtw * 16 + lo) * 512 + kc * 32 + hi * 16;
      v16h a = cat8(*(const v8h*)ap, *(const v8h*)(ap + 8));
      const f16* bbase = bs + (nhf * 16 * 32 + lane) * 16;
      v16h bf = *(const v16h*)bbase;
#pragma unroll
      for (int nt = 0; nt < 15; ++nt) {
        v16h bn = *(const v16h*)(bbase + (nt + 1) * 512);
        acc[nt] = wmma16(a, bf, acc[nt]);
        bf = bn;
      }
      acc[15] = wmma16(a, bf, acc[15]);
    }
#pragma unroll
    for (int nt = 0; nt < 16; ++nt) {
      int ncol = (nhf * 16 + nt) * 16 + lo;
      float bias = bo[ncol];
#pragma unroll
      for (int r = 0; r < 8; ++r) {
        int row = q0 + mtw * 16 + r + hi * 8;
        out[(size_t)(n * QL + row) * DM + ncol] = acc[nt][r] + bias;
      }
    }
  }
}

extern "C" void kernel_launch(void* const* d_in, const int* in_sizes, int n_in,
                              void* d_out, int out_size, void* d_ws,
                              size_t ws_size, hipStream_t stream) {
  const float* query = (const float*)d_in[0];
  const float* ctx   = (const float*)d_in[1];
  const float* Wq    = (const float*)d_in[2];
  const float* Wk    = (const float*)d_in[3];
  const float* Wv    = (const float*)d_in[4];
  const float* Wo    = (const float*)d_in[5];
  const float* bo    = (const float*)d_in[6];
  f16* ws = (f16*)d_ws;
  float* out = (float*)d_out;

  prep_weights<<<(WQB_SZ + WOB_SZ) / 256, 256, 0, stream>>>(Wq, Wo, ws);
  prep_kv<<<NB * NH, 320, 0, stream>>>(ctx, Wk, Wv, ws);
  attn_main<<<dim3(QL / 64, NB), 256, LDS_TOTAL, stream>>>(query, bo, ws, out);
}
